// model_57595511439650
// MI455X (gfx1250) — matrix operand order repaired and checked by reading, not compiled, not run on hardware
//
#include <hip/hip_runtime.h>
#include <hip/hip_bf16.h>

typedef __attribute__((ext_vector_type(16))) __bf16 v16bf;
typedef __attribute__((ext_vector_type(8)))  float  v8f;

// Vanilla tanh-RNN, fused: input projection + recurrence + output head.
// One wave32 per 16 batch rows. H=64 -> 4 N-tiles of 16; K=64 -> 2 K-blocks of 32.
// Projection and recurrence matmuls via v_wmma_f32_16x16x32_bf16 with f32 accumulation; output head is scalar f32.
// Software-pipelined: projection WMMAs of x_{t+1} overlap tanh/LDS of step t,
// so the serial recurrence chain is only 2 WMMAs deep.

__device__ __forceinline__ float fast_tanh(float z) {
#if __has_builtin(__builtin_amdgcn_tanhf)
    return __builtin_amdgcn_tanhf(z);            // v_tanh_f32 (CDNA5 TRANS op)
#else
    const float e2 = __expf(2.0f * z);           // v_exp_f32
    return 1.0f - 2.0f * __builtin_amdgcn_rcpf(e2 + 1.0f);
#endif
}

__global__ __launch_bounds__(32)
void rnn_fused_kernel(const float* __restrict__ x,
                      const float* __restrict__ wih,   // [64,28]
                      const float* __restrict__ whh,   // [64,64]
                      const float* __restrict__ bih,   // [64]
                      const float* __restrict__ bhh,   // [64]
                      const float* __restrict__ wout,  // [10,64]
                      const float* __restrict__ bout,  // [10]
                      float* __restrict__ out)         // [4096,10]
{
    constexpr int T = 512, I = 28, H = 64, O = 10;

    // h tile for this wave, row-major [m=16][k=64], bf16 (C->A relayout buffer)
    __shared__ __bf16 hbuf[16 * H];

    const int lane    = threadIdx.x;        // 0..31
    const int rowbase = blockIdx.x * 16;    // 16 batch rows per wave
    const int m  = lane & 15;
    const int hi = lane >> 4;               // 0 or 1

    // A-fragment K starts within a 32-wide K block (ISA 7.12.2, 16-bit A 16x32):
    // lanes 0-15: elems 0..7 -> K=0..7,  elems 8..15 -> K=16..23
    // lanes16-31: elems 0..7 -> K=8..15, elems 8..15 -> K=24..31
    const int ks1 = hi ? 8  : 0;
    const int ks2 = hi ? 24 : 16;

    // B-fragment (32x16): col n = lane&15 ; same K order as A (elems 0..7 -> ks1.., 8..15 -> ks2..)
    const int bk0 = hi * 16;

    // ---- Build constant B fragments once (kept in VGPRs for all 512 steps) ----
    // xw = x @ W_ih^T  => B[k][n] = W_ih[n][k], K padded 28 -> 32
    // hh = h @ W_hh^T  => B[k][n] = W_hh[n][k]
    v16bf Bih[4];
    v16bf Bhh[4][2];
    #pragma unroll
    for (int nt = 0; nt < 4; ++nt) {
        const int n = nt * 16 + m;
        #pragma unroll
        for (int e = 0; e < 16; ++e) {
            const int k = (e < 8) ? ks1 + e : ks2 + e - 8;
            Bih[nt][e] = (k < I) ? (__bf16)wih[n * I + k] : (__bf16)0.0f;
        }
        #pragma unroll
        for (int kf = 0; kf < 2; ++kf) {
            #pragma unroll
            for (int e = 0; e < 16; ++e) {
                const int k = kf * 32 + ((e < 8) ? ks1 + e : ks2 + e - 8);
                Bhh[nt][kf][e] = (__bf16)whh[n * H + k];
            }
        }
    }

    // Per-lane bias, broadcast into live v8f registers (C layout col j = nt*16+m,
    // identical for all 8 rows a lane holds). Used directly as WMMA C operand.
    v8f biasC[4];
    #pragma unroll
    for (int nt = 0; nt < 4; ++nt) {
        const int j = nt * 16 + m;
        const float b = bih[j] + bhh[j];
        #pragma unroll
        for (int v = 0; v < 8; ++v) biasC[nt][v] = b;
    }

    // h0 = 0
    v16bf hA[2];
    #pragma unroll
    for (int e = 0; e < 16; ++e) { hA[0][e] = (__bf16)0.0f; hA[1][e] = (__bf16)0.0f; }

    const float* xrow = x + (size_t)(rowbase + m) * T * I;
    const float  tailmask = hi ? 0.0f : 1.0f;   // K=28..31 pad (hi-lane elems 12..15)

    // x_t fragment: 4 aligned float4 loads per lane (row stride 112B, 16B aligned).
    // lo lanes: f4 idx {0,1,4,5} (K 0..7, 16..23); hi lanes: {2,3,6,dummy} (K 8..15, 24..27).
    // Two running byte pointers; +16/+64 fold into the instruction immediate offset.
    const char* p0 = (const char*)xrow + (hi ? 2 : 0) * 16;
    const char* p3 = (const char*)xrow + (hi ? 2 : 5) * 16;

    float4 a0, a1, a2, a3;
    auto loadx = [&]() {
        a0 = *(const float4*)(p0);
        a1 = *(const float4*)(p0 + 16);
        a2 = *(const float4*)(p0 + 64);
        a3 = *(const float4*)(p3);
    };
    auto cvtx = [&]() -> v16bf {
        v16bf f;
        f[0]  = (__bf16)a0.x;  f[1]  = (__bf16)a0.y;  f[2]  = (__bf16)a0.z;  f[3]  = (__bf16)a0.w;
        f[4]  = (__bf16)a1.x;  f[5]  = (__bf16)a1.y;  f[6]  = (__bf16)a1.z;  f[7]  = (__bf16)a1.w;
        f[8]  = (__bf16)a2.x;  f[9]  = (__bf16)a2.y;  f[10] = (__bf16)a2.z;  f[11] = (__bf16)a2.w;
        f[12] = (__bf16)(a3.x * tailmask);  f[13] = (__bf16)(a3.y * tailmask);
        f[14] = (__bf16)(a3.z * tailmask);  f[15] = (__bf16)(a3.w * tailmask);
        return f;
    };

    // ---- Pipeline prologue: cx = bias + x_0 @ W_ih^T ; prefetch x_1 ----
    loadx();
    v8f cx[4];
    {
        const v16bf xA = cvtx();
        #pragma unroll
        for (int nt = 0; nt < 4; ++nt)
            cx[nt] = __builtin_amdgcn_wmma_f32_16x16x32_bf16(false, xA, false, Bih[nt],
                                                             (short)0, biasC[nt], false, false);
    }
    p0 += 112; p3 += 112;
    loadx();                                    // x_1 in flight

    for (int t = 0; t < T; ++t) {
        // ---- Recurrence chain (only 2 WMMAs deep on h): c = cx + h @ W_hh^T ----
        v8f c[4];
        #pragma unroll
        for (int nt = 0; nt < 4; ++nt) {
            v8f cc;
            cc = __builtin_amdgcn_wmma_f32_16x16x32_bf16(false, hA[0], false, Bhh[nt][0], (short)0, cx[nt], false, false);
            cc = __builtin_amdgcn_wmma_f32_16x16x32_bf16(false, hA[1], false, Bhh[nt][1], (short)0, cc,     false, false);
            c[nt] = cc;
        }

        // ---- Off-critical-path: project x_{t+1}, prefetch x_{t+2} (clamped) ----
        {
            const v16bf xA = cvtx();
            #pragma unroll
            for (int nt = 0; nt < 4; ++nt)
                cx[nt] = __builtin_amdgcn_wmma_f32_16x16x32_bf16(false, xA, false, Bih[nt],
                                                                 (short)0, biasC[nt], false, false);
        }
        const int adv = (t + 2 < T) ? 112 : 0;  // uniform; avoids OOB at sequence end
        p0 += adv; p3 += adv;
        loadx();

        // ---- tanh + store to LDS in C layout (col j = nt*16+m, row = v + 8*hi) ----
        #pragma unroll
        for (int nt = 0; nt < 4; ++nt) {
            const int j = nt * 16 + m;
            #pragma unroll
            for (int v = 0; v < 8; ++v) {
                hbuf[(v + 8 * hi) * H + j] = (__bf16)fast_tanh(c[nt][v]);
            }
        }

        // ---- Gather next-step A fragments of h (contiguous 8-elem runs -> ds_read_b128).
        // Single-wave WG: LDS ops execute in order within the wave; the store->load RAW
        // through hbuf is a may-alias dependency the compiler preserves (no barrier needed).
        #pragma unroll
        for (int kf = 0; kf < 2; ++kf) {
            const int base = kf * 32;
            #pragma unroll
            for (int e = 0; e < 8; ++e) {
                hA[kf][e]     = hbuf[m * H + base + ks1 + e];
                hA[kf][e + 8] = hbuf[m * H + base + ks2 + e];
            }
        }
    }

    // ---- Output head: out = h_last @ W_out^T + b_out  (16x10 per wave) ----
    for (int i = lane; i < 16 * O; i += 32) {
        const int r = i / O;
        const int o = i % O;
        float acc = bout[o];
        #pragma unroll
        for (int k = 0; k < H; ++k)
            acc += (float)hbuf[r * H + k] * wout[o * H + k];
        out[(size_t)(rowbase + r) * O + o] = acc;
    }
}

extern "C" void kernel_launch(void* const* d_in, const int* in_sizes, int n_in,
                              void* d_out, int out_size, void* d_ws, size_t ws_size,
                              hipStream_t stream) {
    (void)in_sizes; (void)n_in; (void)d_ws; (void)ws_size; (void)out_size;
    const float* x    = (const float*)d_in[0];
    const float* wih  = (const float*)d_in[1];
    const float* whh  = (const float*)d_in[2];
    const float* bih  = (const float*)d_in[3];
    const float* bhh  = (const float*)d_in[4];
    const float* wout = (const float*)d_in[5];
    const float* bout = (const float*)d_in[6];
    float* out = (float*)d_out;

    constexpr int B = 4096;
    dim3 grid(B / 16);   // 256 single-wave workgroups (one per 16-row strip)
    dim3 block(32);      // wave32
    rnn_fused_kernel<<<grid, block, 0, stream>>>(x, wih, whh, bih, bhh, wout, bout, out);
}
